// ScalableModelManager_58042188038537
// MI455X (gfx1250) — matrix operand order repaired and checked by reading, not compiled, not run on hardware
//
#include <hip/hip_runtime.h>

typedef unsigned short u16;
typedef __attribute__((ext_vector_type(16))) __bf16 v16bf;
typedef __attribute__((ext_vector_type(8)))  float  v8f;
typedef __attribute__((ext_vector_type(4)))  unsigned int v4u;

union Frag { v4u q[2]; v16bf v; };

__device__ __forceinline__ u16 f2bf(float f) {
  unsigned u = __builtin_bit_cast(unsigned, f);
  u += 0x7FFFu + ((u >> 16) & 1u);
  return (u16)(u >> 16);
}
__device__ __forceinline__ float bf2f(u16 h) {
  unsigned u = ((unsigned)h) << 16;
  return __builtin_bit_cast(float, u);
}
__device__ __forceinline__ float sigmoidf_(float x) { return 1.f / (1.f + __expf(-x)); }

// ---------------------------------------------------------------------------
// CDNA5 async global->LDS 16B copy (ASYNCcnt-tracked, no VGPR round trip).
// LDS operand = low 32 bits of the generic pointer (LDS aperture truncation).
// ---------------------------------------------------------------------------
__device__ __forceinline__ void async_copy16(const u16* gsrc, u16* lds) {
  unsigned ldsa = (unsigned)(size_t)lds;
  asm volatile("global_load_async_to_lds_b128 %0, %1, off"
               :: "v"(ldsa), "v"(gsrc) : "memory");
}
__device__ __forceinline__ void async_wait0() {
  asm volatile("s_wait_asynccnt 0" ::: "memory");
}

// ---------------------------------------------------------------------------
// Generic bf16 WMMA GEMM: C[M,N] = act(A[M,K] * B(+transB) + bias), batched (z)
// 128 threads = 4 waves; each wave owns 32 rows x 64 cols -> 8 wmma / K-step.
// Double-buffered LDS: async loads for tile k+1 overlap WMMAs on tile k.
// BM=128, BN=64, BK=32 (one wmma K-step)
// ---------------------------------------------------------------------------
#define BM 128
#define BN 64
#define BK 32
#define AST 40   // LDS row stride (halfs) for A tile, padded vs 32
#define BST 40   // LDS row stride (halfs) for B tile (stored [n][k])

template<int ACT, bool OUT_BF16, bool TRANSB>
__global__ __launch_bounds__(128) void gemm_kernel(
    const u16* __restrict__ A, long lda,
    const u16* __restrict__ B, long ldb, long strideB,
    const float* __restrict__ bias, long strideBias,
    void* __restrict__ Cv, long ldc, long strideC, int K)
{
  __shared__ __align__(16) u16 As[2][BM * AST];
  __shared__ __align__(16) u16 Bs[2][BN * BST];
  const int tid = threadIdx.x;
  const int bm = blockIdx.y, bn = blockIdx.x, bz = blockIdx.z;
  const u16* Bp = B + (long)bz * strideB;
  const float* biasp = bias + (long)bz * strideBias;

  const int lane = tid & 31;
  const int lr = lane & 15;
  const bool hi = lane >= 16;
  const int m0 = (tid >> 5) * 32;       // 4 waves x 32 rows

  v8f acc[8] = {};                      // [mh*4 + nt]

  // stage tile at K-offset kt into LDS buffer `buf`
  auto stage = [&](int kt, int buf) {
    // A tile (128x32 halfs): 4 x 16B async per thread
#pragma unroll
    for (int it = 0; it < 4; ++it) {
      int idx = it * 128 + tid;
      int row = idx >> 2, c8 = (idx & 3) * 8;
      async_copy16(A + (long)(bm * BM + row) * lda + kt + c8, &As[buf][row * AST + c8]);
    }
    if (TRANSB) {
      // B row-major [N][K]; straight async copy into Bs[n][k]
#pragma unroll
      for (int it = 0; it < 2; ++it) {
        int idx = it * 128 + tid;
        int n = idx >> 2, c8 = (idx & 3) * 8;
        async_copy16(Bp + (long)(bn * BN + n) * ldb + kt + c8, &Bs[buf][n * BST + c8]);
      }
    } else {
      // B row-major [K][N]; transpose on store (VGPR path, DScnt-tracked)
#pragma unroll
      for (int it = 0; it < 2; ++it) {
        int idx = it * 128 + tid;
        int kk = idx >> 3, n0 = (idx & 7) * 8;
        const u16* src = Bp + (long)(kt + kk) * ldb + bn * BN + n0;
        v4u val = *(const v4u*)src;
        const u16* vh = (const u16*)&val;
#pragma unroll
        for (int j = 0; j < 8; ++j) Bs[buf][(n0 + j) * BST + kk] = vh[j];
        if (kt + BK < K) __builtin_prefetch(src + BK * ldb, 0, 1);
      }
    }
  };

  // ---- pipelined main loop ----
  stage(0, 0);
  async_wait0();
  __syncthreads();
  int cur = 0;
  for (int kt = 0; kt < K; kt += BK) {
    if (kt + BK < K) stage(kt + BK, cur ^ 1);   // prefetch next tile (other buffer)

    // ---- A fragments (2 x 16x32), per documented bf16 A layout ----
    Frag a0, a1;
    {
      int ab = hi ? 8 : 0;              // lanes>=16 hold K {8..15, 24..31}
      int r0 = m0 + lr, r1 = m0 + 16 + lr;
      a0.q[0] = *(const v4u*)&As[cur][r0 * AST + ab];
      a0.q[1] = *(const v4u*)&As[cur][r0 * AST + 16 + ab];
      a1.q[0] = *(const v4u*)&As[cur][r1 * AST + ab];
      a1.q[1] = *(const v4u*)&As[cur][r1 * AST + 16 + ab];
    }
    int kb = hi ? 8 : 0;                // B: lanes>=16 hold K {8..15, 24..31}
#pragma unroll
    for (int nt = 0; nt < 4; ++nt) {
      Frag b;
      int n = nt * 16 + lr;
      b.q[0] = *(const v4u*)&Bs[cur][n * BST + kb];
      b.q[1] = *(const v4u*)&Bs[cur][n * BST + kb + 16];
      acc[nt] = __builtin_amdgcn_wmma_f32_16x16x32_bf16(
          false, a0.v, false, b.v, (short)0, acc[nt], false, false);
      acc[4 + nt] = __builtin_amdgcn_wmma_f32_16x16x32_bf16(
          false, a1.v, false, b.v, (short)0, acc[4 + nt], false, false);
    }

    async_wait0();      // next tile has landed in LDS
    __syncthreads();    // all waves done reading cur + next tile visible
    cur ^= 1;
  }

  // ---- epilogue: C VGPR r -> M = m0 + mh*16 + r + (hi?8:0), N = nt*16 + lr ----
  u16* Cb = (u16*)Cv + (long)bz * strideC;
  float* Cf = (float*)Cv + (long)bz * strideC;
#pragma unroll
  for (int mh = 0; mh < 2; ++mh) {
#pragma unroll
    for (int nt = 0; nt < 4; ++nt) {
      long col = (long)bn * BN + nt * 16 + lr;
      float bv = biasp[col];
#pragma unroll
      for (int r = 0; r < 8; ++r) {
        long row = (long)bm * BM + m0 + mh * 16 + r + (hi ? 8 : 0);
        float x = acc[mh * 4 + nt][r] + bv;
        if (ACT == 1) x = fmaxf(x, 0.f);
        if (OUT_BF16) Cb[row * ldc + col] = f2bf(x);
        else          Cf[row * ldc + col] = x;
      }
    }
  }
}

// ---------------------------------------------------------------------------
// fp32 -> bf16 conversion
// ---------------------------------------------------------------------------
__global__ void cvt_kernel(const float* __restrict__ in, u16* __restrict__ out, long n) {
  long i = (long)blockIdx.x * blockDim.x + threadIdx.x;
  if (i < n) out[i] = f2bf(in[i]);
}

__global__ void zero_kernel(unsigned* __restrict__ p, long n) {
  long i = (long)blockIdx.x * blockDim.x + threadIdx.x;
  if (i < n) p[i] = 0u;
}

// fill x_concat[:, :, 1024:1280] with asset embeddings (broadcast over S=64)
__global__ void fill_emb_kernel(const int* __restrict__ ids, const float* __restrict__ emb,
                                u16* __restrict__ xcat) {
  int idx = blockIdx.x * 256 + threadIdx.x;   // 512*256
  int na = idx >> 8, j = idx & 255;
  u16 b = f2bf(emb[(long)ids[na] * 256 + j]);
  u16* dst = xcat + (long)na * 64 * 1280 + 1024 + j;
#pragma unroll 4
  for (int s = 0; s < 64; ++s) dst[(long)s * 1280] = b;
}

// LSTM pointwise gate math for one timestep. gate order: i,f,g,o (pytorch)
__global__ void lstm_step_kernel(const float* __restrict__ xg, const float* __restrict__ g,
                                 float* __restrict__ c, u16* __restrict__ hcur,
                                 u16* __restrict__ hseq, int t) {
  int idx = blockIdx.x * 256 + threadIdx.x;   // 512*512
  int na = idx >> 9, j = idx & 511;
  long xoff = (long)(na * 64 + t) * 2048 + j;
  long goff = (long)na * 2048 + j;
  float gi = xg[xoff]        + g[goff];
  float gf = xg[xoff + 512]  + g[goff + 512];
  float gg = xg[xoff + 1024] + g[goff + 1024];
  float go = xg[xoff + 1536] + g[goff + 1536];
  float cn = sigmoidf_(gf) * c[idx] + sigmoidf_(gi) * tanhf(gg);
  float h  = sigmoidf_(go) * tanhf(cn);
  c[idx] = cn;
  u16 hb = f2bf(h);
  hcur[idx] = hb;
  hseq[(long)(na * 64 + t) * 512 + j] = hb;
}

// ---------------------------------------------------------------------------
// Cross-asset attention at final timestep only.
// One block per (batch b, head h): q,k,v tiles are 64(assets) x 32(dk).
// scores = q k^T / sqrt(32) via wmma; softmax over assets; out = att @ v.
// ---------------------------------------------------------------------------
__global__ __launch_bounds__(256) void attention_kernel(
    const u16* __restrict__ q, const u16* __restrict__ k,
    const u16* __restrict__ v, u16* __restrict__ o)
{
  __shared__ __align__(16) u16 qs[64 * 40];
  __shared__ __align__(16) u16 ks[64 * 40];
  __shared__ __align__(16) u16 vs[64 * 40];
  __shared__ __align__(16) float sc[64 * 65];
  __shared__ __align__(16) u16 att[64 * 72];

  const int tid = threadIdx.x;
  const int b = blockIdx.x >> 4, h = blockIdx.x & 15;
  {
    int a = tid >> 2, c8 = (tid & 3) * 8;
    long src = (long)(b * 64 + a) * 512 + h * 32 + c8;
    async_copy16(&q[src], &qs[a * 40 + c8]);
    async_copy16(&k[src], &ks[a * 40 + c8]);
    async_copy16(&v[src], &vs[a * 40 + c8]);
  }
  async_wait0();
  __syncthreads();

  const int lane = tid & 31, lr = lane & 15;
  const bool hi = lane >= 16;
  const int wv = tid >> 5;

  // ---- scores: 16 tiles of 16x16, 2 per wave. B = k (Bt layout = k[a'][dk]) ----
#pragma unroll
  for (int rep = 0; rep < 2; ++rep) {
    int tile = wv + rep * 8;
    int mi = tile >> 2, ni = tile & 3;
    Frag a, bf;
    int ab = hi ? 8 : 0;
    a.q[0] = *(const v4u*)&qs[(mi * 16 + lr) * 40 + ab];
    a.q[1] = *(const v4u*)&qs[(mi * 16 + lr) * 40 + 16 + ab];
    int kb = hi ? 8 : 0;
    bf.q[0] = *(const v4u*)&ks[(ni * 16 + lr) * 40 + kb];
    bf.q[1] = *(const v4u*)&ks[(ni * 16 + lr) * 40 + kb + 16];
    v8f acc = {};
    acc = __builtin_amdgcn_wmma_f32_16x16x32_bf16(false, a.v, false, bf.v, (short)0, acc, false, false);
#pragma unroll
    for (int r = 0; r < 8; ++r)
      sc[(mi * 16 + r + (hi ? 8 : 0)) * 65 + ni * 16 + lr] = acc[r] * 0.17677669529663687f;
  }
  __syncthreads();

  // ---- softmax over assets (columns), one row per thread ----
  if (tid < 64) {
    float mx = -1e30f;
    for (int j = 0; j < 64; ++j) mx = fmaxf(mx, sc[tid * 65 + j]);
    float sum = 0.f;
    for (int j = 0; j < 64; ++j) sum += __expf(sc[tid * 65 + j] - mx);
    float inv = 1.f / sum;
    for (int j = 0; j < 64; ++j) att[tid * 72 + j] = f2bf(__expf(sc[tid * 65 + j] - mx) * inv);
  }
  __syncthreads();

  // ---- out = att(64x64) @ v(64x32): 8 tiles, one per wave, K=64 (2 steps) ----
  {
    int mi = wv >> 1, ni = wv & 1;
    v8f acc = {};
#pragma unroll
    for (int kt = 0; kt < 64; kt += 32) {
      Frag a, bf;
      int ab = kt + (hi ? 8 : 0);
      a.q[0] = *(const v4u*)&att[(mi * 16 + lr) * 72 + ab];
      a.q[1] = *(const v4u*)&att[(mi * 16 + lr) * 72 + ab + 16];
      u16* bh = (u16*)&bf;
      int n = ni * 16 + lr;
#pragma unroll
      for (int i = 0; i < 16; ++i)
        bh[i] = vs[(kt + (hi ? 8 : 0) + i + (i >= 8 ? 8 : 0)) * 40 + n];
      acc = __builtin_amdgcn_wmma_f32_16x16x32_bf16(false, a.v, false, bf.v, (short)0, acc, false, false);
    }
#pragma unroll
    for (int r = 0; r < 8; ++r) {
      long row = (long)b * 64 + mi * 16 + r + (hi ? 8 : 0);
      o[row * 512 + h * 32 + ni * 16 + lr] = f2bf(acc[r]);
    }
  }
}

// ---------------------------------------------------------------------------
// Ensemble head second layer: (512x256) @ w2(256x1) + b2, activation.
// act: 0=tanh(pos), 1=sigmoid(conf), 2=softplus(vol). One block per head m.
// ---------------------------------------------------------------------------
__global__ __launch_bounds__(256) void head_gemv_kernel(
    const u16* __restrict__ h1, const float* __restrict__ w2,
    const float* __restrict__ b2, float* __restrict__ out, int act)
{
  int m = blockIdx.x;
  const float* wp = w2 + (long)m * 256;
  for (int r = threadIdx.x; r < 512; r += 256) {
    const u16* hp = h1 + ((long)m * 512 + r) * 256;
    float s = 0.f;
    for (int kk = 0; kk < 256; ++kk) s += bf2f(hp[kk]) * wp[kk];
    s += b2[m];
    float y;
    if (act == 0)      y = tanhf(s);
    else if (act == 1) y = sigmoidf_(s);
    else               y = (s > 20.f) ? s : log1pf(__expf(s));
    out[(long)m * 512 + r] = y;
  }
}

// ---------------------------------------------------------------------------
template<int ACT, bool OB, bool TB>
static void launch_gemm(hipStream_t s, const u16* A, long lda, const u16* B, long ldb, long sB,
                        const float* bias, long sBias, void* C, long ldc, long sC,
                        int M, int N, int K, int batch) {
  dim3 grid(N / BN, M / BM, batch);
  gemm_kernel<ACT, OB, TB><<<grid, 128, 0, s>>>(A, lda, B, ldb, sB, bias, sBias, C, ldc, sC, K);
}

extern "C" void kernel_launch(void* const* d_in, const int* in_sizes, int n_in,
                              void* d_out, int out_size, void* d_ws, size_t ws_size,
                              hipStream_t stream) {
  (void)in_sizes; (void)n_in; (void)out_size; (void)ws_size;
  const long NROW = 32768;                  // B*A*S

  // ---- workspace bump allocator ----
  size_t off = 0;
  char* wsb = (char*)d_ws;
  auto alloc = [&](size_t bytes) -> void* {
    off = (off + 255) & ~(size_t)255;
    void* p = wsb + off;
    off += bytes;
    return p;
  };

  u16* enc_w1b = (u16*)alloc(1024L * 2048 * 2);
  u16* enc_w2b = (u16*)alloc(2048L * 1024 * 2);
  u16* wih0b   = (u16*)alloc(2048L * 1280 * 2);
  u16* wihb    = (u16*)alloc(3L * 2048 * 512 * 2);   // layers 1..3
  u16* whhb    = (u16*)alloc(4L * 2048 * 512 * 2);   // layers 0..3
  u16* wqb     = (u16*)alloc(512L * 512 * 2);
  u16* wkb     = (u16*)alloc(512L * 512 * 2);
  u16* wvb     = (u16*)alloc(512L * 512 * 2);
  u16* wob     = (u16*)alloc(512L * 512 * 2);
  u16* hw1b    = (u16*)alloc(3L * 50 * 512 * 256 * 2);
  // slot0: [features_bf 64MB | enc1 128MB] later reused as xg (fp32, 256MB)
  char* slot0  = (char*)alloc(268435456);
  u16*   feat_b = (u16*)slot0;
  u16*   enc1   = (u16*)(slot0 + 67108864);
  float* xg     = (float*)slot0;
  u16* xcat  = (u16*)alloc(NROW * 1280 * 2);
  u16* hseqA = (u16*)alloc(NROW * 512 * 2);
  u16* hseqB = (u16*)alloc(NROW * 512 * 2);
  float* gbuf = (float*)alloc(512L * 2048 * 4);
  float* ccur = (float*)alloc(512L * 512 * 4);
  u16* hcur  = (u16*)alloc(512L * 512 * 2);
  u16* qbuf  = (u16*)alloc(512L * 512 * 2);
  u16* kbuf  = (u16*)alloc(512L * 512 * 2);
  u16* vbuf  = (u16*)alloc(512L * 512 * 2);
  u16* obuf  = (u16*)alloc(512L * 512 * 2);
  u16* finalb = (u16*)alloc(512L * 512 * 2);
  u16* h1buf = (u16*)alloc(50L * 512 * 256 * 2);

  auto cvt = [&](const void* in, u16* out, long n) {
    cvt_kernel<<<dim3((unsigned)((n + 255) / 256)), 256, 0, stream>>>((const float*)in, out, n);
  };
  auto zero = [&](void* p, long bytes) {
    long n = bytes / 4;
    zero_kernel<<<dim3((unsigned)((n + 255) / 256)), 256, 0, stream>>>((unsigned*)p, n);
  };

  // ---- weight / input conversions to bf16 ----
  cvt(d_in[0],  feat_b, 33554432);
  cvt(d_in[2],  enc_w1b, 1024L * 2048);
  cvt(d_in[4],  enc_w2b, 2048L * 1024);
  cvt(d_in[7],  wih0b,   2048L * 1280);
  cvt(d_in[11], wihb,    3L * 2048 * 512);
  cvt(d_in[8],  whhb,            2048L * 512);
  cvt(d_in[12], whhb + 2048L * 512, 3L * 2048 * 512);
  cvt(d_in[15], wqb, 512L * 512);
  cvt(d_in[17], wkb, 512L * 512);
  cvt(d_in[19], wvb, 512L * 512);
  cvt(d_in[21], wob, 512L * 512);
  cvt(d_in[23], hw1b,                     50L * 512 * 256);
  cvt(d_in[27], hw1b + 50L * 512 * 256,   50L * 512 * 256);
  cvt(d_in[31], hw1b + 100L * 512 * 256,  50L * 512 * 256);

  // ---- encoder: relu(X @ W1 + b1), relu(. @ W2 + b2) -> xcat[:, :1024] ----
  launch_gemm<1, true, false>(stream, feat_b, 1024, enc_w1b, 2048, 0,
      (const float*)d_in[3], 0, enc1, 2048, 0, (int)NROW, 2048, 1024, 1);
  launch_gemm<1, true, false>(stream, enc1, 2048, enc_w2b, 1024, 0,
      (const float*)d_in[5], 0, xcat, 1280, 0, (int)NROW, 1024, 2048, 1);
  fill_emb_kernel<<<512, 256, 0, stream>>>((const int*)d_in[1], (const float*)d_in[6], xcat);

  // ---- 4-layer LSTM ----
  u16* hin = nullptr;
  u16* hout = hseqA;
  for (int l = 0; l < 4; ++l) {
    if (l == 0) {
      // xg = xcat @ wih0^T + bih0   (wih0 is [2048,1280] row-major -> transB)
      launch_gemm<0, false, true>(stream, xcat, 1280, wih0b, 1280, 0,
          (const float*)d_in[9], 0, xg, 2048, 0, (int)NROW, 2048, 1280, 1);
    } else {
      launch_gemm<0, false, true>(stream, hin, 512, wihb + (long)(l - 1) * 2048 * 512, 512, 0,
          (const float*)d_in[13] + (l - 1) * 2048, 0, xg, 2048, 0, (int)NROW, 2048, 512, 1);
    }
    zero(hcur, 512L * 512 * 2);
    zero(ccur, 512L * 512 * 4);
    const u16* whh_l = whhb + (long)l * 2048 * 512;
    const float* bhh = (l == 0) ? (const float*)d_in[10]
                                : (const float*)d_in[14] + (l - 1) * 2048;
    for (int t = 0; t < 64; ++t) {
      // g = h @ whh^T + bhh   (512 x 2048 x 512, transB)
      launch_gemm<0, false, true>(stream, hcur, 512, whh_l, 512, 0,
          bhh, 0, gbuf, 2048, 0, 512, 2048, 512, 1);
      lstm_step_kernel<<<1024, 256, 0, stream>>>(xg, gbuf, ccur, hcur, hout, t);
    }
    hin = hout;
    hout = (hout == hseqA) ? hseqB : hseqA;
  }
  u16* hfin = hin;   // (na, s, 512) bf16

  // ---- attention at s = S-1 only (output depends only on last timestep) ----
  const u16* a63 = hfin + 63L * 512;   // row na at stride 64*512
  launch_gemm<0, true, false>(stream, a63, 64L * 512, wqb, 512, 0,
      (const float*)d_in[16], 0, qbuf, 512, 0, 512, 512, 512, 1);
  launch_gemm<0, true, false>(stream, a63, 64L * 512, wkb, 512, 0,
      (const float*)d_in[18], 0, kbuf, 512, 0, 512, 512, 512, 1);
  launch_gemm<0, true, false>(stream, a63, 64L * 512, wvb, 512, 0,
      (const float*)d_in[20], 0, vbuf, 512, 0, 512, 512, 512, 1);
  attention_kernel<<<128, 256, 0, stream>>>(qbuf, kbuf, vbuf, obuf);
  launch_gemm<0, true, false>(stream, obuf, 512, wob, 512, 0,
      (const float*)d_in[22], 0, finalb, 512, 0, 512, 512, 512, 1);

  // ---- 50-way ensemble heads (pos/conf/vol) ----
  float* outp = (float*)d_out;
  for (int hd = 0; hd < 3; ++hd) {
    const float* b1 = (const float*)d_in[24 + hd * 4];
    const float* w2 = (const float*)d_in[25 + hd * 4];
    const float* b2 = (const float*)d_in[26 + hd * 4];
    launch_gemm<1, true, false>(stream, finalb, 512, hw1b + (long)hd * 50 * 512 * 256, 256,
        512L * 256, b1, 256, h1buf, 256, 512L * 256, 512, 256, 512, 50);
    head_gemv_kernel<<<50, 256, 0, stream>>>(h1buf, w2, b2, outp + (long)hd * 25600, hd);
  }
}
